// GRUModel_78554951844380
// MI455X (gfx1250) — matrix operand order repaired and checked by reading, not compiled, not run on hardware
//
#include <hip/hip_runtime.h>
#include <hip/hip_bf16.h>

// ---------------------------------------------------------------------------
// GRUCell with fixed h0, vectorized over T (reference never updates hid):
//   gi = X[131072,256] @ w_ih^T            -> bf16 WMMA (v_wmma_f32_16x16x32_bf16)
//   gh = hid @ w_hh^T + b_hh  (64x768)     -> tiny, precomputed once
//   out = (1-z)*tanh(i_n + r*h_n) + z*hid  -> fused epilogue on WMMA acc layout
//
// Per block: 64 rows of X, 512 threads (16 waves).
//  - x tile converted to bf16 in LDS (32 KiB), read 8x per wave as A-fragments
//  - weights double-buffered per-K-chunk in LDS (2 x 48 KiB) via ASYNC
//    global->LDS copies (ASYNCcnt + s_wait_asynccnt), overlapping WMMA compute
//  - wave (rt, jg) computes 4 j-tiles x 3 gates = 12 acc tiles; gates combined
//    in registers, sigmoid/tanh/blend fused, out + out[-1] stored directly.
// L2 weight traffic: 2048 blocks x 384 KiB = 768 MiB (4x less than 16-row tiles).
// ---------------------------------------------------------------------------

#define T_DIM   2048
#define B_DIM   64
#define H_DIM   256
#define TB_ROWS (T_DIM * B_DIM)      // 131072
#define G3      (3 * H_DIM)          // 768

typedef __attribute__((ext_vector_type(16))) __bf16        v16bf;
typedef __attribute__((ext_vector_type(8)))  float         v8f;
typedef __attribute__((ext_vector_type(4)))  unsigned int  v4u;
typedef __attribute__((ext_vector_type(4)))  int           v4i;

// Exactly-typed pointers for the async global->LDS builtin (from diagnostics:
// param0 is a non-const v4i pointer in the global (__device__) address space).
typedef __attribute__((address_space(1))) v4i* gas_v4i_p;
typedef __attribute__((address_space(3))) v4i* las_v4i_p;

union Frag32 {            // 32 bytes = 16 bf16 = one WMMA A/B fragment per lane
    v4u   q[2];
    v16bf b;
};

__device__ __forceinline__ unsigned short f2bf(float f) {
    unsigned u = __builtin_bit_cast(unsigned, f);
    u += 0x7FFFu + ((u >> 16) & 1u);          // round-to-nearest-even
    return (unsigned short)(u >> 16);
}

// --- prep 1: convert w_ih (768x256 f32, row-major) -> bf16 in workspace -----
__global__ __launch_bounds__(256) void gru_conv_w(const float* __restrict__ w,
                                                  unsigned short* __restrict__ wbf) {
    int i = blockIdx.x * 256 + threadIdx.x;   // grid 768 -> 196608 threads
    wbf[i] = f2bf(w[i]);
}

// --- prep 2: gh[b][g] = b_hh[g] + sum_k hid[b][k] * w_hh[g][k]  (64x768) ----
__global__ __launch_bounds__(256) void gru_gh(const float* __restrict__ hid,
                                              const float* __restrict__ w_hh,
                                              const float* __restrict__ b_hh,
                                              float* __restrict__ ghd) {
    int gid = blockIdx.x * 256 + threadIdx.x; // grid 192 -> 49152 = 64*768
    int b = gid & 63;
    int g = gid >> 6;
    const float* hr = hid  + b * H_DIM;
    const float* wr = w_hh + g * H_DIM;
    float s = b_hh[g];
    #pragma unroll 8
    for (int k = 0; k < H_DIM; ++k) s += hr[k] * wr[k];
    ghd[b * G3 + g] = s;
}

// Async copy of weight K-slice c (768 rows x 32 k, bf16) into LDS.
// 3072 x 16B lane-copies; 6 per thread with 512 threads. Tracked by ASYNCcnt.
__device__ __forceinline__ void issue_slice_copy(const unsigned short* __restrict__ wbf,
                                                 int c, unsigned short* dstLds, int tid) {
    #pragma unroll
    for (int it = 0; it < 6; ++it) {
        int f    = tid + it * 512;          // 0..3071
        int row  = f >> 2;                  // weight row 0..767
        int part = f & 3;                   // 16B quarter of the 64B row-slice
        const unsigned short* gsrc = wbf + row * H_DIM + c * 32 + part * 8;
        unsigned short*       ldst = dstLds + row * 32 + part * 8;
#if __has_builtin(__builtin_amdgcn_global_load_async_to_lds_b128)
        __builtin_amdgcn_global_load_async_to_lds_b128(
            (gas_v4i_p)gsrc, (las_v4i_p)ldst, 0, 0);
#else
        unsigned lds32 = (unsigned)(unsigned long long)ldst;   // LDS byte address
        asm volatile("global_load_async_to_lds_b128 %0, %1, off"
                     :: "v"(lds32), "v"(gsrc) : "memory");
#endif
    }
}

// --- main: 64 rows of X per block; wave = (row-tile, j-group) ---------------
__global__ __launch_bounds__(512) void gru_main(const float* __restrict__ x,
                                                const unsigned short* __restrict__ wbf,
                                                const float* __restrict__ ghd,
                                                const float* __restrict__ hid,
                                                const float* __restrict__ b_ih,
                                                float* __restrict__ out,
                                                float* __restrict__ out_last) {
    __shared__ __align__(16) unsigned short ldsA[64 * 256];      // 32 KiB bf16 x-tile
    __shared__ __align__(16) unsigned short ldsB[2][G3 * 32];    // 2 x 48 KiB weight slices

    const int tid     = threadIdx.x;
    const int rowBase = blockIdx.x * 64;
    const int wave    = tid >> 5;        // 0..15
    const int lane    = tid & 31;
    const int m       = lane & 15;
    const int hi      = lane >> 4;
    const int rt      = wave & 3;        // row-tile 0..3 (16 rows each)
    const int jg      = wave >> 2;       // j-group 0..3 (4 j-tiles each)

    // Kick off async weight slice 0 while we stage x.
    issue_slice_copy(wbf, 0, ldsB[0], tid);

    // Stage x[rowBase:+64, :] -> bf16 LDS (coalesced f32x2 loads).
    {
        const float* xr = x + (size_t)rowBase * H_DIM;
        #pragma unroll
        for (int i = 0; i < 16; ++i) {
            int idx = i * 1024 + tid * 2;
            float2 v = *(const float2*)(xr + idx);
            unsigned p = (unsigned)f2bf(v.x) | ((unsigned)f2bf(v.y) << 16);
            *(unsigned*)(&ldsA[idx]) = p;
        }
    }

    v8f acc[4][3] = {};   // [jj][gate] : 16x16 f32 tiles, 96 VGPRs

    #pragma unroll
    for (int c = 0; c < 8; ++c) {               // K chunks of 32
        const int buf = c & 1;
        __builtin_amdgcn_s_wait_asynccnt(0);    // slice c landed (this wave's copies)
        __syncthreads();                        // everyone's copies + x staging visible
        if (c < 7)                              // overlap: fetch slice c+1 now
            issue_slice_copy(wbf, c + 1, ldsB[buf ^ 1], tid);

        // A fragment (16x32 bf16): lanes 0-15 K {0..7,16..23}, lanes 16-31 K {8..15,24..31}
        Frag32 a;
        {
            const v4u* pa = (const v4u*)&ldsA[(rt * 16 + m) * 256 + c * 32 + hi * 8];
            a.q[0] = pa[0];
            a.q[1] = pa[2];
        }
        #pragma unroll
        for (int jj = 0; jj < 4; ++jj) {
            const int jcol = (jg * 4 + jj) * 16;
            #pragma unroll
            for (int g = 0; g < 3; ++g) {
                // B fragment (32x16 bf16): lane n = column, lanes 0-15 K {0..7,16..23}, lanes 16-31 K {8..15,24..31}
                const int grow = g * H_DIM + jcol + m;            // weight row
                Frag32 bfr;
                const v4u* pb = (const v4u*)&ldsB[buf][grow * 32 + hi * 8];
                bfr.q[0] = pb[0];
                bfr.q[1] = pb[2];
                acc[jj][g] = __builtin_amdgcn_wmma_f32_16x16x32_bf16(
                    false, a.b, false, bfr.b, (short)0, acc[jj][g], false, false);
            }
        }
    }

    // Fused GRU epilogue on accumulator layout: (vgpr i, lane l) -> M=i+8*hi, N=l&15.
    #pragma unroll
    for (int jj = 0; jj < 4; ++jj) {
        const int col = (jg * 4 + jj) * 16 + m;
        const float bir = b_ih[col];
        const float biz = b_ih[col + 256];
        const float bin = b_ih[col + 512];
        #pragma unroll
        for (int i = 0; i < 8; ++i) {
            const int row = rowBase + rt * 16 + i + hi * 8;
            const int b   = row & 63;                 // row = t*64 + b
            const float* ghr = ghd + b * G3;
            const float hr = ghr[col];
            const float hz = ghr[col + 256];
            const float hn = ghr[col + 512];
            const float hv = hid[b * H_DIM + col];

            const float rr = acc[jj][0][i] + bir + hr;
            const float zz = acc[jj][1][i] + biz + hz;
            const float nn = acc[jj][2][i] + bin;

            const float r = 1.0f / (1.0f + __expf(-rr));
            const float z = 1.0f / (1.0f + __expf(-zz));
            const float n = tanhf(nn + r * hn);
            const float o = (1.0f - z) * n + z * hv;

            out[(size_t)row * H_DIM + col] = o;
            if (row >= TB_ROWS - B_DIM)               // out[-1] duplicate
                out_last[(row - (TB_ROWS - B_DIM)) * H_DIM + col] = o;
        }
    }
}

extern "C" void kernel_launch(void* const* d_in, const int* in_sizes, int n_in,
                              void* d_out, int out_size, void* d_ws, size_t ws_size,
                              hipStream_t stream) {
    const float* x    = (const float*)d_in[0];
    const float* hid  = (const float*)d_in[1];
    const float* w_ih = (const float*)d_in[2];
    const float* w_hh = (const float*)d_in[3];
    const float* b_ih = (const float*)d_in[4];
    const float* b_hh = (const float*)d_in[5];

    // workspace: [0, 384KiB) bf16 w_ih ; [384KiB, 576KiB) gh f32 64x768
    unsigned short* wbf = (unsigned short*)d_ws;
    float* ghd = (float*)((char*)d_ws + (size_t)G3 * H_DIM * sizeof(unsigned short));

    float* out      = (float*)d_out;
    float* out_last = out + (size_t)TB_ROWS * H_DIM;

    gru_conv_w<<<G3, 256, 0, stream>>>(w_ih, wbf);
    gru_gh<<<(B_DIM * G3) / 256, 256, 0, stream>>>(hid, w_hh, b_hh, ghd);
    gru_main<<<TB_ROWS / 64, 512, 0, stream>>>(x, wbf, ghd, hid, b_ih, out, out_last);
}
